// DecisionVQVAE_1116691497623
// MI455X (gfx1250) — compile-verified, matrix operand order checked by reading, not run on hardware
//
#include <hip/hip_runtime.h>

// ---------------------------------------------------------------------------
// Types for CDNA5 WMMA (wave32, 16x16x32 bf16 -> f32)
// ---------------------------------------------------------------------------
typedef __bf16 bf16;
typedef __attribute__((ext_vector_type(16))) __bf16 v16bf;
typedef __attribute__((ext_vector_type(8)))  __bf16 bf16x8;
typedef __attribute__((ext_vector_type(8)))  float  v8f;
typedef __attribute__((ext_vector_type(4)))  __bf16 bf16x4;
typedef __attribute__((ext_vector_type(4)))  unsigned int u32x4;
typedef __attribute__((ext_vector_type(4)))  int i32x4v;

typedef __attribute__((address_space(1))) i32x4v g1_i32x4;  // global <4 x i32>
typedef __attribute__((address_space(3))) i32x4v l3_i32x4;  // LDS    <4 x i32>
typedef void __attribute__((address_space(3))) lds_void;

#if defined(__has_builtin)
#if __has_builtin(__builtin_amdgcn_global_load_async_to_lds_b128)
#define USE_ASYNC_BUILTIN 1
#endif
#if __has_builtin(__builtin_amdgcn_s_wait_asynccnt)
#define USE_WAITA_BUILTIN 1
#endif
#endif

// One 16-byte async fill: global -> LDS, tracked by ASYNCcnt (no VGPR transit).
__device__ __forceinline__ void async_copy_b128(void* lds, const void* gl) {
#ifdef USE_ASYNC_BUILTIN
  __builtin_amdgcn_global_load_async_to_lds_b128(
      (g1_i32x4*)(void*)gl, (l3_i32x4*)lds, 0, 0);
#else
  unsigned loff = (unsigned)(unsigned long long)(lds_void*)lds;
  asm volatile("global_load_async_to_lds_b128 %0, %1, off"
               :: "v"(loff), "v"(gl) : "memory");
#endif
}

__device__ __forceinline__ void wait_async0() {
#ifdef USE_WAITA_BUILTIN
  __builtin_amdgcn_s_wait_asynccnt(0);
#else
  asm volatile("s_wait_asynccnt 0x0" ::: "memory");
#endif
}

// A lane's 16 fragment elements are two contiguous 8-element K-runs:
// k in [8*hf, 8*hf+8) and [16+8*hf, 16+8*hf+8)  (ISA 16-bit A/B layout).
// With K-contiguous LDS storage each run is one ds_load_b128.
__device__ __forceinline__ v16bf load_frag(const bf16* row, int hf) {
  bf16x8 lo = *(const bf16x8*)(row + 8 * hf);
  bf16x8 hi = *(const bf16x8*)(row + 16 + 8 * hf);
  return __builtin_shufflevector(lo, hi, 0, 1, 2, 3, 4, 5, 6, 7,
                                 8, 9, 10, 11, 12, 13, 14, 15);
}

// ---------------------------------------------------------------------------
// Prep kernels
// ---------------------------------------------------------------------------
__global__ void cvt_f32_bf16(const float* __restrict__ s, bf16* __restrict__ d, int n) {
  int i = blockIdx.x * blockDim.x + threadIdx.x;
  int st = gridDim.x * blockDim.x;
  for (; i < n; i += st) d[i] = (bf16)s[i];
}

// W[K,N] f32 -> WT[N,K] bf16 (N-major so GEMM LDS staging stays vectorized)
__global__ void cvt_T(const float* __restrict__ w, bf16* __restrict__ wt, int K, int N) {
  int i = blockIdx.x * blockDim.x + threadIdx.x;  // i = k*N + n (coalesced read)
  if (i < K * N) {
    int k = i / N, n = i % N;
    wt[(size_t)n * K + k] = (bf16)w[i];
  }
}

// one thread per code: ||c||^2; also zero loss partials
__global__ void csq_kernel(const float* __restrict__ cb, float* __restrict__ csq,
                           float* __restrict__ part) {
  int code = threadIdx.x;  // <<<1,512>>>
  float s = 0.f;
  for (int c = 0; c < 256; ++c) { float v = cb[code * 256 + c]; s += v * v; }
  csq[code] = s;
  part[code] = 0.f;
}

// ---------------------------------------------------------------------------
// Tiled GEMM: C[M,N] = act(A[M,K] * B + bias), B given as BT[N,K] bf16.
// block = 256 threads (8 waves), tile M=128, N=128, K-step=32.
// Async global->LDS fills, ping-pong buffers, one barrier per K-step.
// ---------------------------------------------------------------------------
template <bool A_F32, bool RELU, bool OUT_F32>
__global__ __launch_bounds__(256) void gemm_kernel(
    const void* __restrict__ Aptr, const bf16* __restrict__ BT,
    const float* __restrict__ bias, void* __restrict__ Cptr, int K, int N) {
  __shared__ bf16 As[2][128 * 40];  // [m][k] pitch 40 -> conflict-free b128 frags
  __shared__ bf16 Bs[2][128 * 40];  // [n][k] pitch 40

  const int tid = threadIdx.x;
  const int lane = tid & 31, w = tid >> 5;
  const int hf = lane >> 4, lr = lane & 15;
  const int m0 = blockIdx.x * 128;
  const int n0 = blockIdx.y * 128;
  const int T = K >> 5;

  const bf16* Abf = (const bf16*)Aptr + (A_F32 ? 0 : (size_t)m0 * K);
  const bf16* Brow = BT + (size_t)n0 * K;

  // async-stage a 128(row) x 32(k) bf16 tile, K-contiguous rows, pitch 40
  auto stage_async = [&](bf16* dst, const bf16* src, int k0) {
#pragma unroll
    for (int i = 0; i < 2; ++i) {
      int v = tid + 256 * i;
      int r = v >> 2, c8 = (v & 3) << 3;
      async_copy_b128(&dst[r * 40 + c8], src + (size_t)r * K + k0 + c8);
    }
  };

  const v8f zero8 = {0.f, 0.f, 0.f, 0.f, 0.f, 0.f, 0.f, 0.f};
  v8f acc[8] = {zero8, zero8, zero8, zero8, zero8, zero8, zero8, zero8};

  // prologue: fill buffer 0
  if (!A_F32) stage_async(As[0], Abf, 0);
  stage_async(Bs[0], Brow, 0);

  for (int kt = 0; kt < T; ++kt) {
    const int cur = kt & 1;
    wait_async0();       // my fills for buf[cur] landed
    __syncthreads();     // everyone's landed; everyone done reading buf[cur^1]
    if (kt + 1 < T) {    // overlap next tile's fills with this tile's math
      if (!A_F32) stage_async(As[cur ^ 1], Abf, (kt + 1) * 32);
      stage_async(Bs[cur ^ 1], Brow, (kt + 1) * 32);
    }
    if (A_F32) {
      // layer-1 only: f32 -> bf16 convert requires VALU, stage synchronously
      const float* A = (const float*)Aptr;
      const int k0 = kt * 32;
#pragma unroll
      for (int i = 0; i < 4; ++i) {
        int v = tid + 256 * i;
        int r = v >> 3, c4 = (v & 7) << 2;
        float4 f = *(const float4*)(A + (size_t)(m0 + r) * K + k0 + c4);
        bf16x4 o;
        o[0] = (bf16)f.x; o[1] = (bf16)f.y; o[2] = (bf16)f.z; o[3] = (bf16)f.w;
        *(bf16x4*)&As[cur][r * 40 + c4] = o;
      }
      __syncthreads();
    }

    // fragments + WMMA (wave w: rows [16w,16w+16) x all 128 cols)
    const int m = w * 16 + lr;
    v16bf a = load_frag(&As[cur][m * 40], hf);
#pragma unroll
    for (int t = 0; t < 8; ++t) {
      v16bf b = load_frag(&Bs[cur][(t * 16 + lr) * 40], hf);
      acc[t] = __builtin_amdgcn_wmma_f32_16x16x32_bf16(
          false, a, false, b, (short)0, acc[t], false, false);
    }
  }

  // epilogue: bias (+ReLU), store f32 or bf16
#pragma unroll
  for (int t = 0; t < 8; ++t) {
    int gn = n0 + t * 16 + lr;
    float bv = bias[gn];
#pragma unroll
    for (int r = 0; r < 8; ++r) {
      int gm = m0 + w * 16 + r + 8 * hf;
      float v = acc[t][r] + bv;
      if (RELU) v = v > 0.f ? v : 0.f;
      if (OUT_F32)
        ((float*)Cptr)[(size_t)gm * N + gn] = v;
      else
        ((bf16*)Cptr)[(size_t)gm * N + gn] = (bf16)v;
    }
  }
}

// ---------------------------------------------------------------------------
// VQ kernel: distances via WMMA, fused argmin, q gather, commit-loss partial.
// 256 threads / 8 waves / 128 tokens. z tile async-staged into LDS (~66KB of
// CDNA5's 320KB WGP LDS); A fragments hoisted into registers for the whole
// 512-code sweep; codebook chunks async-staged.
// ---------------------------------------------------------------------------
__global__ __launch_bounds__(256) void vq_kernel(
    const bf16* __restrict__ zb, const bf16* __restrict__ CBb,
    const float* __restrict__ cbf, const float* __restrict__ csq,
    int* __restrict__ idx_out, bf16* __restrict__ qb,
    float* __restrict__ part) {
  extern __shared__ char smem[];
  bf16* Az = (bf16*)smem;                                    // 128 x 264 [m][k]
  bf16* Bt = (bf16*)(smem + 128 * 264 * 2);                  // 64 x 264  [n][k]
  int* idxBuf = (int*)(smem + 128 * 264 * 2 + 64 * 264 * 2); // 128
  float* wsum = (float*)(idxBuf + 128);                      // 8

  const int tid = threadIdx.x;
  const int lane = tid & 31, w = tid >> 5;
  const int hf = lane >> 4, lr = lane & 15;
  const int m0 = blockIdx.x * 128;

  // async-stage the whole 128x256 z tile (bf16) into LDS
#pragma unroll
  for (int i = 0; i < 16; ++i) {
    int v = tid + 256 * i;
    int r = v >> 5, c8 = (v & 31) << 3;
    async_copy_b128(&Az[r * 264 + c8], zb + (size_t)(m0 + r) * 256 + c8);
  }
  wait_async0();
  __syncthreads();

  // hoist all 8 A fragments (K=256) into registers for the full code sweep
  const int m = w * 16 + lr;
  v16bf afrag[8];
#pragma unroll
  for (int ks = 0; ks < 8; ++ks) afrag[ks] = load_frag(&Az[m * 264 + ks * 32], hf);

  float bestv[8];
  int besti[8];
#pragma unroll
  for (int r = 0; r < 8; ++r) { bestv[r] = 3.4e38f; besti[r] = 0; }

  const v8f zero8 = {0.f, 0.f, 0.f, 0.f, 0.f, 0.f, 0.f, 0.f};

  for (int ch = 0; ch < 8; ++ch) {  // 8 chunks x 64 codes = 512 codes
    __syncthreads();  // previous chunk's compute done
    // async-stage 64(n) x 256(k) codebook chunk (row-major == N-major B)
#pragma unroll
    for (int i = 0; i < 8; ++i) {
      int v = tid + 256 * i;
      int n = v >> 5, c8 = (v & 31) << 3;
      async_copy_b128(&Bt[n * 264 + c8],
                      CBb + (size_t)(ch * 64 + n) * 256 + c8);
    }
    wait_async0();
    __syncthreads();

    v8f acc[4] = {zero8, zero8, zero8, zero8};
#pragma unroll
    for (int ks = 0; ks < 8; ++ks) {  // K = 256 = 8 x 32
#pragma unroll
      for (int t = 0; t < 4; ++t) {
        v16bf b = load_frag(&Bt[(t * 16 + lr) * 264 + ks * 32], hf);
        acc[t] = __builtin_amdgcn_wmma_f32_16x16x32_bf16(
            false, afrag[ks], false, b, (short)0, acc[t], false, false);
      }
    }
    // fused argmin epilogue: score = ||c||^2 - 2*dot
#pragma unroll
    for (int t = 0; t < 4; ++t) {
      int ng = ch * 64 + t * 16 + lr;
      float cq = csq[ng];
#pragma unroll
      for (int r = 0; r < 8; ++r) {
        float s = cq - 2.f * acc[t][r];
        if (s < bestv[r]) { bestv[r] = s; besti[r] = ng; }
      }
    }
  }

  // reduce argmin across the 16 lanes holding each row (ties -> lowest index)
#pragma unroll
  for (int r = 0; r < 8; ++r) {
#pragma unroll
    for (int off = 8; off >= 1; off >>= 1) {
      float ov = __shfl_xor(bestv[r], off, 32);
      int oi = __shfl_xor(besti[r], off, 32);
      if (ov < bestv[r] || (ov == bestv[r] && oi < besti[r])) {
        bestv[r] = ov; besti[r] = oi;
      }
    }
  }
  if (lr == 0) {
#pragma unroll
    for (int r = 0; r < 8; ++r) {
      int row = w * 16 + r + 8 * hf;
      idx_out[m0 + row] = besti[r];
      idxBuf[row] = besti[r];
    }
  }
  __syncthreads();

  // gather q (bf16) + accumulate commitment loss (z - q)^2
  float ls = 0.f;
  const int d0 = lane * 8;
  for (int rr = 0; rr < 16; ++rr) {
    int row = w * 16 + rr;
    int code = idxBuf[row];
    u32x4 qq = *(const u32x4*)(CBb + (size_t)code * 256 + d0);
    *(u32x4*)(qb + (size_t)(m0 + row) * 256 + d0) = qq;
    const bf16* zrow = &Az[row * 264 + d0];
    const float* crow = cbf + (size_t)code * 256 + d0;
#pragma unroll
    for (int j = 0; j < 8; ++j) {
      float dz = (float)zrow[j] - crow[j];
      ls += dz * dz;
    }
  }
#pragma unroll
  for (int off = 16; off >= 1; off >>= 1) ls += __shfl_xor(ls, off, 32);
  if (lane == 0) wsum[w] = ls;
  __syncthreads();
  if (tid == 0) {  // deterministic per-block partial (no float atomics)
    float s = 0.f;
    for (int i = 0; i < 8; ++i) s += wsum[i];
    part[blockIdx.x] = s;
  }
}

__global__ void loss_final(const float* __restrict__ part, float* __restrict__ out) {
  if (threadIdx.x == 0 && blockIdx.x == 0) {
    float s = 0.f;
    for (int i = 0; i < 512; ++i) s += part[i];
    out[0] = s * (1.0f / (65536.0f * 256.0f));  // COMMIT_W = 1.0
  }
}

// ---------------------------------------------------------------------------
// Launcher
// ---------------------------------------------------------------------------
extern "C" void kernel_launch(void* const* d_in, const int* in_sizes, int n_in,
                              void* d_out, int out_size, void* d_ws, size_t ws_size,
                              hipStream_t stream) {
  (void)in_sizes; (void)n_in; (void)out_size; (void)ws_size;
  const float* x  = (const float*)d_in[0];
  const float* W1 = (const float*)d_in[1];
  const float* b1 = (const float*)d_in[2];
  const float* W2 = (const float*)d_in[3];
  const float* b2 = (const float*)d_in[4];
  const float* cb = (const float*)d_in[5];
  const float* W3 = (const float*)d_in[6];
  const float* b3 = (const float*)d_in[7];
  const float* W4 = (const float*)d_in[8];
  const float* b4 = (const float*)d_in[9];

  char* ws = (char*)d_ws;
  constexpr size_t O_W1  = 0;                                  // WT1 [512,768]
  constexpr size_t O_W2  = O_W1 + (size_t)768 * 512 * 2;       // WT2 [256,512]
  constexpr size_t O_W3  = O_W2 + (size_t)512 * 256 * 2;       // WT3 [512,256]
  constexpr size_t O_W4  = O_W3 + (size_t)256 * 512 * 2;       // WT4 [768,512]
  constexpr size_t O_CBB = O_W4 + (size_t)512 * 768 * 2;       // codebook bf16 [512,256]
  constexpr size_t O_CSQ = O_CBB + (size_t)512 * 256 * 2;
  constexpr size_t O_PRT = O_CSQ + 512 * 4;
  constexpr size_t O_H   = O_PRT + 512 * 4;
  constexpr size_t O_Z   = O_H + (size_t)65536 * 512 * 2;
  constexpr size_t O_Q   = O_Z + (size_t)65536 * 256 * 2;
  constexpr size_t O_H2  = O_Q + (size_t)65536 * 256 * 2;

  bf16* W1t  = (bf16*)(ws + O_W1);
  bf16* W2t  = (bf16*)(ws + O_W2);
  bf16* W3t  = (bf16*)(ws + O_W3);
  bf16* W4t  = (bf16*)(ws + O_W4);
  bf16* CBb  = (bf16*)(ws + O_CBB);
  float* csqp = (float*)(ws + O_CSQ);
  float* prtp = (float*)(ws + O_PRT);
  bf16* hb   = (bf16*)(ws + O_H);
  bf16* zbuf = (bf16*)(ws + O_Z);
  bf16* qb   = (bf16*)(ws + O_Q);
  bf16* h2b  = (bf16*)(ws + O_H2);

  float* recon    = (float*)d_out;
  int* idx_out    = (int*)d_out + (size_t)65536 * 768;
  float* loss_out = (float*)d_out + (size_t)65536 * 768 + 65536;

  cvt_T<<<384, 256, 0, stream>>>(W1, W1t, 768, 512);
  cvt_T<<<128, 256, 0, stream>>>(W2, W2t, 512, 256);
  cvt_T<<<128, 256, 0, stream>>>(W3, W3t, 256, 512);
  cvt_T<<<384, 256, 0, stream>>>(W4, W4t, 512, 768);
  cvt_f32_bf16<<<128, 256, 0, stream>>>(cb, CBb, 512 * 256);
  csq_kernel<<<1, 512, 0, stream>>>(cb, csqp, prtp);

  // encoder
  gemm_kernel<true,  true,  false><<<dim3(512, 4), 256, 0, stream>>>(x,  W1t, b1, hb,   768, 512);
  gemm_kernel<false, false, false><<<dim3(512, 2), 256, 0, stream>>>(hb, W2t, b2, zbuf, 512, 256);

  // VQ: distances + argmin + gather + loss partials
  size_t vq_smem = (size_t)128 * 264 * 2 + (size_t)64 * 264 * 2 + 128 * 4 + 8 * 4;
  vq_kernel<<<512, 256, vq_smem, stream>>>(zbuf, CBb, cb, csqp, idx_out, qb, prtp);

  // decoder
  gemm_kernel<false, true,  false><<<dim3(512, 4), 256, 0, stream>>>(qb,  W3t, b3, h2b,  256, 512);
  gemm_kernel<false, false, true ><<<dim3(512, 6), 256, 0, stream>>>(h2b, W4t, b4, recon, 512, 768);

  loss_final<<<1, 1, 0, stream>>>(prtp, loss_out);
}
